// UTG_40458591928547
// MI455X (gfx1250) — matrix operand order repaired and checked by reading, not compiled, not run on hardware
//
#include <hip/hip_runtime.h>
#include <cstdint>

// ---------------------------------------------------------------------------
// Transformer-XL layer forward for MI455X (gfx1250, wave32, WMMA bf16,
// Tensor Data Mover stage-in for the GEMM tiles).
// ---------------------------------------------------------------------------

#define QLEN_   1024
#define BSZ_    4
#define DMODEL_ 1024
#define NHEAD_  16
#define DHEAD_  64
#define DINNER_ 4096
#define MROWS_  (QLEN_*BSZ_)   // 4096 flattened (i,b) rows

typedef __attribute__((ext_vector_type(16))) __bf16 v16bf;
typedef __attribute__((ext_vector_type(8)))  float  v8f;

// TDM descriptor groups (6-arg builtin on this toolchain:
//   (uint32x4 g0, int32x8 g1, int32x4, int32x4, int32x8, i32 cpol))
typedef __attribute__((ext_vector_type(4))) unsigned int tdm_v4u;
typedef __attribute__((ext_vector_type(8))) int          tdm_v8i;
typedef __attribute__((ext_vector_type(4))) int          tdm_v4i;

union Frag32B { int4 q[2]; v16bf v; };

__device__ __forceinline__ unsigned short f2bf(float f) {
  unsigned int u = __float_as_uint(f);
  u += 0x7FFFu + ((u >> 16) & 1u);          // round-to-nearest-even
  return (unsigned short)(u >> 16);
}

// ---------------------------------------------------------------------------
// TDM: 2-D bf16 tile load, global -> LDS, with per-row padding:
//   tile is tile_w elems (x 2B) wide, tile_h rows; row-in-tensor stride = K.
//   pad_interval=16 dwords (64B = one 32-elem row), pad_amount=4 dwords (16B)
//   => LDS row stride 80B (16B aligned, bank-spread).
// Issued by one wave; completion tracked with TENSORcnt.
// ---------------------------------------------------------------------------
__device__ __forceinline__ void tdm_load_tile_2d(unsigned lds_addr,
                                                 const void* gptr,
                                                 int row_stride_elems,   // K
                                                 int tensor_rows,        // M/N
                                                 int tile_w, int tile_h)
{
  unsigned long long ga = (unsigned long long)(uintptr_t)gptr;

  tdm_v4u g0;
  g0[0] = 1u;                                              // count=1 (valid), user mode
  g0[1] = lds_addr;                                        // LDS byte address
  g0[2] = (unsigned)(ga & 0xFFFFFFFFu);                    // global_addr[31:0]
  g0[3] = (unsigned)((ga >> 32) & 0x01FFFFFFu) | (2u << 30); // addr[56:32] | type=2

  tdm_v8i g1;
  g1[0] = (1 << 16)        // data_size = 1 -> 2 bytes
        | (1 << 20)        // pad_enable
        | (3 << 22)        // pad_interval: 16 dwords
        | (3 << 25);       // pad_amount : 4 dwords
  g1[1] = (row_stride_elems & 0xFFFF) << 16;               // tensor_dim0[15:0]
  g1[2] = ((unsigned)row_stride_elems >> 16)
        | ((tensor_rows & 0xFFFF) << 16);                  // tdim0 hi | tdim1 lo
  g1[3] = ((unsigned)tensor_rows >> 16) | (tile_w << 16);  // tdim1 hi | tile_dim0
  g1[4] = tile_h & 0xFFFF;                                 // tile_dim1 | tile_dim2=0
  g1[5] = row_stride_elems;                                // tensor_dim0_stride lo
  g1[6] = 0;                                               // stride hi | dim1_stride lo
  g1[7] = 0;

  tdm_v4i g2;
  g2[0] = 1; g2[1] = 1; g2[2] = 0; g2[3] = 0;              // tdim2=1, tdim3=1, tile_dim3=0
  tdm_v4i g3;
  g3[0] = 0; g3[1] = 0; g3[2] = 0; g3[3] = 0;
  tdm_v8i g4;                                              // unused trailing group
  g4[0] = 0; g4[1] = 0; g4[2] = 0; g4[3] = 0;
  g4[4] = 0; g4[5] = 0; g4[6] = 0; g4[7] = 0;

  __builtin_amdgcn_tensor_load_to_lds(g0, g1, g2, g3, g4, 0);
}

// ---------------------------------------------------------------------------
// elementwise f32 -> bf16 pack
// ---------------------------------------------------------------------------
__global__ void pack_bf16_kernel(const float* __restrict__ s,
                                 unsigned short* __restrict__ d, long n) {
  long i = (long)blockIdx.x * blockDim.x + threadIdx.x;
  if (i < n) d[i] = f2bf(s[i]);
}

// ---------------------------------------------------------------------------
// GEMM: C[M][N] = A[M][K] * B[N][K]^T  (bf16 in, f32 accum)
// block = 256 threads (8 waves, 2x4). Tile M=128 N=128 K=32.
// Tiles staged by the Tensor Data Mover (wave 0 issues, TENSORcnt sync),
// double-buffered in LDS with 80B row stride (TDM pad).
// ---------------------------------------------------------------------------
#define ROWB_  80                 // LDS bytes per 32-elem row (64 data + 16 pad)
#define TILEB_ (128 * ROWB_)      // 10240 bytes per tile

__global__ __launch_bounds__(256)
void gemm_bf16_nt(const unsigned short* __restrict__ A,
                  const unsigned short* __restrict__ B,
                  const float* __restrict__ bias,        // [N] or null
                  float* __restrict__ Cf,                // [M][N] or null
                  unsigned short* __restrict__ Cb,       // [M][N] or null
                  int M, int N, int K, int relu)
{
  // [buf0: A,B][buf1: A,B] = 40 KB
  __shared__ __align__(16) char smraw[4 * TILEB_];

  const int tid  = threadIdx.x;
  const int lane = tid & 31;
  const int w    = tid >> 5;
  const int wm   = w >> 2;            // 0..1 : 64-row strip
  const int wn   = w & 3;             // 0..3 : 32-col strip
  const int hi   = lane >> 4;
  const int lr   = lane & 15;

  const long rowA0 = (long)blockIdx.y * 128;
  const long rowB0 = (long)blockIdx.x * 128;

  const unsigned ldsBase = (unsigned)(unsigned long long)(uintptr_t)(void*)&smraw[0];

  v8f acc[4][2];
  #pragma unroll
  for (int mi = 0; mi < 4; ++mi)
    #pragma unroll
    for (int ni = 0; ni < 2; ++ni)
      #pragma unroll
      for (int e = 0; e < 8; ++e) acc[mi][ni][e] = 0.0f;

  const int nk = K >> 5;

  // prologue: TDM tile 0 -> buffer 0
  if (w == 0) {
    tdm_load_tile_2d(ldsBase,          A + rowA0 * K, K, M, 32, 128);
    tdm_load_tile_2d(ldsBase + TILEB_, B + rowB0 * K, K, N, 32, 128);
  }

  for (int kt = 0; kt < nk; ++kt) {
    const int cur = kt & 1;
    if (w == 0) __builtin_amdgcn_s_wait_tensorcnt(0);  // tile kt resident in LDS
    __syncthreads();                                   // publish; prior reads done

    if (w == 0 && kt + 1 < nk) {                       // DMA next tile while computing
      const long k0 = (long)(kt + 1) << 5;
      const unsigned dst = ldsBase + (unsigned)(1 - cur) * 2u * TILEB_;
      tdm_load_tile_2d(dst,          A + rowA0 * K + k0, K, M, 32, 128);
      tdm_load_tile_2d(dst + TILEB_, B + rowB0 * K + k0, K, N, 32, 128);
    }

    const char* abase = smraw + (size_t)cur * 2 * TILEB_;
    const char* bbase = abase + TILEB_;

    Frag32B af[4], bfg[2];
    #pragma unroll
    for (int mi = 0; mi < 4; ++mi) {
      const char* p = abase + (wm * 64 + mi * 16 + lr) * ROWB_;
      af[mi].q[0] = *reinterpret_cast<const int4*>(p + hi * 16);       // K 0-7 / 8-15
      af[mi].q[1] = *reinterpret_cast<const int4*>(p + 32 + hi * 16);  // K 16-23 / 24-31
    }
    #pragma unroll
    for (int ni = 0; ni < 2; ++ni) {
      const char* p = bbase + (wn * 32 + ni * 16 + lr) * ROWB_;
      bfg[ni].q[0] = *reinterpret_cast<const int4*>(p + hi * 16);      // K 0-7 / 8-15
      bfg[ni].q[1] = *reinterpret_cast<const int4*>(p + 32 + hi * 16); // K 16-23 / 24-31
    }

    #pragma unroll
    for (int mi = 0; mi < 4; ++mi)
      #pragma unroll
      for (int ni = 0; ni < 2; ++ni)
        acc[mi][ni] = __builtin_amdgcn_wmma_f32_16x16x32_bf16(
            false, af[mi].v, false, bfg[ni].v, (short)0, acc[mi][ni], false, false);
  }

  // epilogue: C layout => lane l, vgpr g : row = g + 8*(l>>4), col = l&15
  #pragma unroll
  for (int mi = 0; mi < 4; ++mi)
    #pragma unroll
    for (int ni = 0; ni < 2; ++ni)
      #pragma unroll
      for (int g = 0; g < 8; ++g) {
        long row = rowA0 + wm * 64 + mi * 16 + g + 8 * hi;
        long col = rowB0 + wn * 32 + ni * 16 + lr;
        float v = acc[mi][ni][g];
        if (bias) v += bias[col];
        if (relu) v = fmaxf(v, 0.0f);
        if (Cf) Cf[row * N + col] = v;
        if (Cb) Cb[row * N + col] = f2bf(v);
      }
}

// ---------------------------------------------------------------------------
// Build attention operands per (b,n):
//   Aq[bn][i][0:64]  = (q + r_w_bias)*scale ; Aq[..][64:128] = (q + r_r_bias)*scale
//   Bk[bn][j][0:64]  = k                    ; Bk[..][64:128] = r_head_k[j,n,:]
//   Vt[bn][d][j]     = v[j,b,n,d]           (B^T layout for P·V)
// ---------------------------------------------------------------------------
__global__ __launch_bounds__(256)
void build_attn_inputs(const float* __restrict__ wheads,   // [4096][3072]
                       const float* __restrict__ rheadk,   // [1024][1024]
                       const float* __restrict__ rwb,      // [16][64]
                       const float* __restrict__ rrb,      // [16][64]
                       unsigned short* __restrict__ Aq,
                       unsigned short* __restrict__ Bk,
                       unsigned short* __restrict__ Vt)
{
  const int bn = blockIdx.y;
  const int b = bn >> 4, n = bn & 15;
  const int i = blockIdx.x * 4 + (threadIdx.x >> 6);
  const int d = threadIdx.x & 63;
  const float scale = 0.125f;   // 1/sqrt(64)

  const long base = ((long)i * BSZ_ + b) * 3072;
  float q  = wheads[base + n * 64 + d];
  float k  = wheads[base + 1024 + n * 64 + d];
  float v  = wheads[base + 2048 + n * 64 + d];
  float rk = rheadk[(long)i * 1024 + n * 64 + d];

  long ab = (long)bn * (1024 * 128) + (long)i * 128;
  Aq[ab + d]      = f2bf((q + rwb[n * 64 + d]) * scale);
  Aq[ab + 64 + d] = f2bf((q + rrb[n * 64 + d]) * scale);
  Bk[ab + d]      = f2bf(k);
  Bk[ab + 64 + d] = f2bf(rk);
  Vt[(long)bn * (64 * 1024) + (long)d * 1024 + i] = f2bf(v);
}

// ---------------------------------------------------------------------------
// Fused causal flash-attention (per (b,n), per 128-row i-block; each wave
// owns 16 query rows, streams 32-wide j tiles, online softmax).
// Output written directly as bf16 in [i*4+b][n*64+d] layout for the O GEMM.
// ---------------------------------------------------------------------------
__global__ __launch_bounds__(256)
void flash_attn(const unsigned short* __restrict__ Aq,   // [64][1024][128]
                const unsigned short* __restrict__ Bk,   // [64][1024][128]
                const unsigned short* __restrict__ Vt,   // [64][64][1024]
                unsigned short* __restrict__ O)          // [4096][1024]
{
  __shared__ int4 pbuf[8][16][4];   // per-wave 16x32 bf16 P tile (swizzled)

  const int tid  = threadIdx.x;
  const int lane = tid & 31;
  const int w    = tid >> 5;
  const int hi   = lane >> 4;
  const int lr   = lane & 15;
  const int bn   = blockIdx.y;
  const int b = bn >> 4, n = bn & 15;
  const int i0 = blockIdx.x * 128 + w * 16;

  const unsigned short* aq = Aq + (long)bn * (1024 * 128);
  const unsigned short* bk = Bk + (long)bn * (1024 * 128);
  const unsigned short* vt = Vt + (long)bn * (64 * 1024);

  // Q fragments (16 rows x 128 K), loaded once, directly in A-frag layout.
  Frag32B qf[4];
  #pragma unroll
  for (int kf = 0; kf < 4; ++kf) {
    const unsigned short* p = aq + (long)(i0 + lr) * 128 + kf * 32 + hi * 8;
    qf[kf].q[0] = *reinterpret_cast<const int4*>(p);
    qf[kf].q[1] = *reinterpret_cast<const int4*>(p + 16);
  }

  v8f oacc[4];
  #pragma unroll
  for (int nf = 0; nf < 4; ++nf)
    #pragma unroll
    for (int e = 0; e < 8; ++e) oacc[nf][e] = 0.0f;

  float mrun[8], lrun[8];
  #pragma unroll
  for (int g = 0; g < 8; ++g) { mrun[g] = -3.0e38f; lrun[g] = 0.0f; }

  unsigned short* pw = reinterpret_cast<unsigned short*>(&pbuf[w][0][0]);
  const int njt = (i0 + 16 + 31) >> 5;

  for (int jt = 0; jt < njt; ++jt) {
    const int j0 = jt * 32;

    v8f s[2];
    #pragma unroll
    for (int nf = 0; nf < 2; ++nf)
      #pragma unroll
      for (int e = 0; e < 8; ++e) s[nf][e] = 0.0f;

    #pragma unroll
    for (int kf = 0; kf < 4; ++kf) {
      Frag32B b0, b1;
      const unsigned short* p0 = bk + (long)(j0 + lr) * 128 + kf * 32 + hi * 8;
      b0.q[0] = *reinterpret_cast<const int4*>(p0);
      b0.q[1] = *reinterpret_cast<const int4*>(p0 + 16);
      const unsigned short* p1 = bk + (long)(j0 + 16 + lr) * 128 + kf * 32 + hi * 8;
      b1.q[0] = *reinterpret_cast<const int4*>(p1);
      b1.q[1] = *reinterpret_cast<const int4*>(p1 + 16);
      s[0] = __builtin_amdgcn_wmma_f32_16x16x32_bf16(false, qf[kf].v, false, b0.v,
                                                     (short)0, s[0], false, false);
      s[1] = __builtin_amdgcn_wmma_f32_16x16x32_bf16(false, qf[kf].v, false, b1.v,
                                                     (short)0, s[1], false, false);
    }

    // causal mask
    #pragma unroll
    for (int nf = 0; nf < 2; ++nf)
      #pragma unroll
      for (int g = 0; g < 8; ++g) {
        int j = j0 + nf * 16 + lr;
        int i = i0 + g + 8 * hi;
        if (j > i) s[nf][g] = -1.0e30f;
      }

    // online softmax update (row stats per accumulator row g+8*hi)
    #pragma unroll
    for (int g = 0; g < 8; ++g) {
      float t = fmaxf(s[0][g], s[1][g]);
      t = fmaxf(t, __shfl_xor(t, 1, 32));
      t = fmaxf(t, __shfl_xor(t, 2, 32));
      t = fmaxf(t, __shfl_xor(t, 4, 32));
      t = fmaxf(t, __shfl_xor(t, 8, 32));
      float mnew = fmaxf(mrun[g], t);
      float sc   = __expf(mrun[g] - mnew);
      float p0   = __expf(s[0][g] - mnew);
      float p1   = __expf(s[1][g] - mnew);
      s[0][g] = p0; s[1][g] = p1;
      float ts = p0 + p1;
      ts += __shfl_xor(ts, 1, 32);
      ts += __shfl_xor(ts, 2, 32);
      ts += __shfl_xor(ts, 4, 32);
      ts += __shfl_xor(ts, 8, 32);
      lrun[g] = lrun[g] * sc + ts;
      mrun[g] = mnew;
      #pragma unroll
      for (int nf = 0; nf < 4; ++nf) oacc[nf][g] *= sc;
    }

    // stage P (bf16) through wave-private LDS to convert C-layout -> A-layout
    #pragma unroll
    for (int nf = 0; nf < 2; ++nf)
      #pragma unroll
      for (int g = 0; g < 8; ++g) {
        int row = g + 8 * hi;
        int e   = nf * 16 + lr;
        int off = row * 32 + (((e >> 3) ^ ((row >> 2) & 3)) << 3) + (e & 7);
        pw[off] = f2bf(s[nf][g]);
      }
    __threadfence_block();   // order same-wave LDS stores before loads

    Frag32B pf;
    {
      int sw = (lr >> 2) & 3;
      pf.q[0] = pbuf[w][lr][hi ^ sw];
      pf.q[1] = pbuf[w][lr][(2 + hi) ^ sw];
    }

    #pragma unroll
    for (int nf = 0; nf < 4; ++nf) {
      Frag32B vf;
      const unsigned short* vp = vt + (long)(nf * 16 + lr) * 1024 + j0 + hi * 8;
      vf.q[0] = *reinterpret_cast<const int4*>(vp);
      vf.q[1] = *reinterpret_cast<const int4*>(vp + 16);
      oacc[nf] = __builtin_amdgcn_wmma_f32_16x16x32_bf16(false, pf.v, false, vf.v,
                                                         (short)0, oacc[nf], false, false);
    }
  }

  // normalize and store bf16 attn_vec rows
  #pragma unroll
  for (int nf = 0; nf < 4; ++nf)
    #pragma unroll
    for (int g = 0; g < 8; ++g) {
      long irow = i0 + g + 8 * hi;
      long col  = (long)n * 64 + nf * 16 + lr;
      O[(irow * BSZ_ + b) * 1024 + col] = f2bf(oacc[nf][g] / lrun[g]);
    }
}

// ---------------------------------------------------------------------------
// out = LayerNorm(X + Y) * gamma + beta  (row length 1024, 256 threads/row)
// ---------------------------------------------------------------------------
__global__ __launch_bounds__(256)
void add_ln_kernel(const float* __restrict__ X, const float* __restrict__ Y,
                   const float* __restrict__ gamma, const float* __restrict__ beta,
                   float* __restrict__ outF, unsigned short* __restrict__ outB)
{
  __shared__ float red[2][8];
  const long row = blockIdx.x;
  const int tid  = threadIdx.x;
  const float* x = X + row * 1024;
  const float* y = Y + row * 1024;

  float v[4];
  float s = 0.0f, s2 = 0.0f;
  #pragma unroll
  for (int k = 0; k < 4; ++k) {
    float t = x[tid + k * 256] + y[tid + k * 256];
    v[k] = t; s += t; s2 += t * t;
  }
  #pragma unroll
  for (int m = 1; m < 32; m <<= 1) {
    s  += __shfl_xor(s,  m, 32);
    s2 += __shfl_xor(s2, m, 32);
  }
  if ((tid & 31) == 0) { red[0][tid >> 5] = s; red[1][tid >> 5] = s2; }
  __syncthreads();
  s  = red[0][tid & 7];
  s2 = red[1][tid & 7];
  #pragma unroll
  for (int m = 1; m < 8; m <<= 1) {
    s  += __shfl_xor(s,  m, 32);
    s2 += __shfl_xor(s2, m, 32);
  }
  const float mu  = s * (1.0f / 1024.0f);
  const float var = s2 * (1.0f / 1024.0f) - mu * mu;
  const float rs  = rsqrtf(var + 1e-5f);
  #pragma unroll
  for (int k = 0; k < 4; ++k) {
    int e = tid + k * 256;
    float o = (v[k] - mu) * rs * gamma[e] + beta[e];
    outF[row * 1024 + e] = o;
    if (outB) outB[row * 1024 + e] = f2bf(o);
  }
}

// ---------------------------------------------------------------------------
// host orchestration
// ---------------------------------------------------------------------------
extern "C" void kernel_launch(void* const* d_in, const int* in_sizes, int n_in,
                              void* d_out, int out_size, void* d_ws, size_t ws_size,
                              hipStream_t stream)
{
  (void)in_sizes; (void)n_in; (void)out_size; (void)ws_size;
  const float* w     = (const float*)d_in[0];
  const float* r     = (const float*)d_in[1];
  // d_in[2] = attn_mask (causal, handled analytically)
  const float* rwb   = (const float*)d_in[3];
  const float* rrb   = (const float*)d_in[4];
  const float* qkvw  = (const float*)d_in[5];
  const float* rnetw = (const float*)d_in[6];
  const float* ow    = (const float*)d_in[7];
  const float* ln1g  = (const float*)d_in[8];
  const float* ln1b  = (const float*)d_in[9];
  const float* ffw1  = (const float*)d_in[10];
  const float* ffb1  = (const float*)d_in[11];
  const float* ffw2  = (const float*)d_in[12];
  const float* ffb2  = (const float*)d_in[13];
  const float* ln2g  = (const float*)d_in[14];
  const float* ln2b  = (const float*)d_in[15];
  float* out = (float*)d_out;

  char* ws = (char*)d_ws;
  size_t off = 0;
  auto alloc = [&](size_t bytes) -> void* {
    void* p = ws + off;
    off += (bytes + 255) & ~(size_t)255;
    return p;
  };

  unsigned short* wb16     = (unsigned short*)alloc((size_t)MROWS_ * DMODEL_ * 2);
  unsigned short* qkvw16   = (unsigned short*)alloc((size_t)3072 * DMODEL_ * 2);
  unsigned short* rnet16   = (unsigned short*)alloc((size_t)DMODEL_ * DMODEL_ * 2);
  unsigned short* r16      = (unsigned short*)alloc((size_t)QLEN_ * DMODEL_ * 2);
  unsigned short* ow16     = (unsigned short*)alloc((size_t)DMODEL_ * DMODEL_ * 2);
  unsigned short* ffw1_16  = (unsigned short*)alloc((size_t)DINNER_ * DMODEL_ * 2);
  unsigned short* ffw2_16  = (unsigned short*)alloc((size_t)DMODEL_ * DINNER_ * 2);
  float*          wheads   = (float*)alloc((size_t)MROWS_ * 3072 * 4);
  float*          rheadk   = (float*)alloc((size_t)QLEN_ * DMODEL_ * 4);
  unsigned short* Aq       = (unsigned short*)alloc((size_t)64 * QLEN_ * 128 * 2);
  unsigned short* Bk       = (unsigned short*)alloc((size_t)64 * QLEN_ * 128 * 2);
  unsigned short* Vt       = (unsigned short*)alloc((size_t)64 * 64 * QLEN_ * 2);
  unsigned short* avec16   = (unsigned short*)alloc((size_t)MROWS_ * DMODEL_ * 2);
  float*          attnout  = (float*)alloc((size_t)MROWS_ * DMODEL_ * 4);
  float*          out1f    = (float*)alloc((size_t)MROWS_ * DMODEL_ * 4);
  unsigned short* out1b    = (unsigned short*)alloc((size_t)MROWS_ * DMODEL_ * 2);
  unsigned short* h16      = (unsigned short*)alloc((size_t)MROWS_ * DINNER_ * 2);
  float*          coref    = (float*)alloc((size_t)MROWS_ * DMODEL_ * 4);

  auto pack = [&](const float* s, unsigned short* d, long n) {
    pack_bf16_kernel<<<dim3((unsigned)((n + 255) / 256)), dim3(256), 0, stream>>>(s, d, n);
  };
  pack(w,     wb16,    (long)MROWS_ * DMODEL_);
  pack(qkvw,  qkvw16,  (long)3072 * DMODEL_);
  pack(rnetw, rnet16,  (long)DMODEL_ * DMODEL_);
  pack(r,     r16,     (long)QLEN_ * DMODEL_);
  pack(ow,    ow16,    (long)DMODEL_ * DMODEL_);
  pack(ffw1,  ffw1_16, (long)DINNER_ * DMODEL_);
  pack(ffw2,  ffw2_16, (long)DMODEL_ * DINNER_);

  // 1) w_heads = w @ qkv_w^T       [4096 x 3072]
  gemm_bf16_nt<<<dim3(3072 / 128, MROWS_ / 128), dim3(256), 0, stream>>>(
      wb16, qkvw16, nullptr, wheads, nullptr, MROWS_, 3072, DMODEL_, 0);

  // 2) r_head_k = r @ r_net_w^T    [1024 x 1024]
  gemm_bf16_nt<<<dim3(DMODEL_ / 128, QLEN_ / 128), dim3(256), 0, stream>>>(
      r16, rnet16, nullptr, rheadk, nullptr, QLEN_, DMODEL_, DMODEL_, 0);

  // 3) build per-head attention operands
  build_attn_inputs<<<dim3(QLEN_ / 4, 64), dim3(256), 0, stream>>>(
      wheads, rheadk, rwb, rrb, Aq, Bk, Vt);

  // 4) fused causal flash attention -> attn_vec (bf16)
  flash_attn<<<dim3(QLEN_ / 128, 64), dim3(256), 0, stream>>>(Aq, Bk, Vt, avec16);

  // 5) attn_out = attn_vec @ o_w^T
  gemm_bf16_nt<<<dim3(DMODEL_ / 128, MROWS_ / 128), dim3(256), 0, stream>>>(
      avec16, ow16, nullptr, attnout, nullptr, MROWS_, DMODEL_, DMODEL_, 0);

  // 6) out1 = LN(w + attn_out)
  add_ln_kernel<<<dim3(MROWS_), dim3(256), 0, stream>>>(
      w, attnout, ln1g, ln1b, out1f, out1b);

  // 7) h = relu(out1 @ ff_w1^T + b1)   (bf16 out only)
  gemm_bf16_nt<<<dim3(DINNER_ / 128, MROWS_ / 128), dim3(256), 0, stream>>>(
      out1b, ffw1_16, ffb1, nullptr, h16, MROWS_, DINNER_, DMODEL_, 1);

  // 8) core = h @ ff_w2^T + b2
  gemm_bf16_nt<<<dim3(DMODEL_ / 128, MROWS_ / 128), dim3(256), 0, stream>>>(
      h16, ffw2_16, ffb2, coref, nullptr, MROWS_, DMODEL_, DINNER_, 0);

  // 9) out = LN(out1 + core)
  add_ln_kernel<<<dim3(MROWS_), dim3(256), 0, stream>>>(
      out1f, coref, ln2g, ln2b, out, nullptr);
}
